// TimeLSTM_12549894439501
// MI455X (gfx1250) — matrix operand order repaired and checked by reading, not compiled, not run on hardware
//
#include <hip/hip_runtime.h>
#include <hip/hip_bf16.h>
#include <math.h>

typedef __bf16 bf16;
typedef __attribute__((ext_vector_type(16))) __bf16 v16bf;
typedef __attribute__((ext_vector_type(8)))  __bf16 v8bf;
typedef __attribute__((ext_vector_type(8)))  float  v8f;

#define TT    8192   // timesteps
#define EMB   1024   // embedding dim
#define HID   2048   // hidden dim
#define G4    8192   // 4*HID (stacked gates i,f,g,o)
#define OUTD  1024   // output dim
#define SCAN_BLOCKS 128
#define FLAG_STRIDE 16   // 64B padding per flag: no false sharing between blocks

// ---------------------------------------------------------------------------
// WMMA helper: D = A(16x32 bf16) * B(32x16 bf16) + C(16x16 f32)
// ---------------------------------------------------------------------------
__device__ __forceinline__ v8f wmma_bf16(v16bf a, v16bf b, v8f c) {
  return __builtin_amdgcn_wmma_f32_16x16x32_bf16(false, a, false, b,
                                                 (short)0, c, false, false);
}

union BV16 { v16bf v; v8bf h[2]; };

// ---------------------------------------------------------------------------
// Prep: f32 -> bf16 conversions, fused bias, zero h-state + publish flags.
// Re-runs every launch so graph replay is deterministic.
// ---------------------------------------------------------------------------
__global__ __launch_bounds__(256) void prep_kernel(
    const float* __restrict__ X, const float* __restrict__ Wih,
    const float* __restrict__ bih, const float* __restrict__ bhh,
    const float* __restrict__ Wlin,
    bf16* __restrict__ Xbf, bf16* __restrict__ Wihbf, bf16* __restrict__ Wlinbf,
    float* __restrict__ bias, float* __restrict__ hbuf, int* __restrict__ flags)
{
  size_t gid    = (size_t)blockIdx.x * blockDim.x + threadIdx.x;
  size_t stride = (size_t)gridDim.x * blockDim.x;
  for (size_t i = gid; i < (size_t)TT * EMB; i += stride)   Xbf[i]    = (bf16)X[i];
  for (size_t i = gid; i < (size_t)G4 * EMB; i += stride)   Wihbf[i]  = (bf16)Wih[i];
  for (size_t i = gid; i < (size_t)OUTD * HID; i += stride) Wlinbf[i] = (bf16)Wlin[i];
  for (size_t i = gid; i < (size_t)G4; i += stride)         bias[i]   = bih[i] + bhh[i];
  for (size_t i = gid; i < (size_t)2 * HID; i += stride)    hbuf[i]   = 0.0f;
  for (size_t i = gid; i < (size_t)SCAN_BLOCKS * FLAG_STRIDE; i += stride) flags[i] = 0;
}

// ---------------------------------------------------------------------------
// bf16 WMMA GEMM:  C[M,N] = A[M,K] * B[N,K]^T + bias[N]   (f32 accumulate)
// One wave computes a 16x64 strip (4 N-tiles): each A tile load feeds 4
// WMMAs. Software-pipelined: tiles for k+32 load into distinct registers
// while the 4 WMMAs for k execute, so load latency hides under matrix work
// (partial s_wait_loadcnt instead of full serialization).
// Per-lane layouts per ISA tables:
//   A (16x32 bf16): lanes 0-15 rows M=lane, K halves {k..k+7, k+16..k+23};
//                   lanes 16-31 same rows, K halves {k+8..k+15, k+24..k+31}.
//   B (32x16 bf16): lane n -> column n, same K halves as A (lo lanes
//                   {k..k+7, k+16..k+23}, hi lanes {k+8..k+15, k+24..k+31}).
//   C (16x16 f32):  element (M = v + 8*(lane>=16), N = lane%16) in acc[v].
// ---------------------------------------------------------------------------
__global__ __launch_bounds__(256) void gemm_bf16_kernel(
    const bf16* __restrict__ A,     // [M, K] row-major
    const bf16* __restrict__ B,     // [N, K] row-major (i.e. B^T of math view)
    const float* __restrict__ bias, // [N]
    float* __restrict__ C,          // [M, N]
    int K, int N)
{
  const int lane = threadIdx.x & 31;
  const int wave = threadIdx.x >> 5;
  const int col  = lane & 15;
  const int hi   = lane >> 4;                     // 0: lanes 0-15, 1: 16-31
  const int t0   = blockIdx.x * 16;               // M tile
  const int r0   = (blockIdx.y * 8 + wave) * 64;  // 4 N-tiles of 16

  const bf16* arow = A + (size_t)(t0 + col) * K + hi * 8;
  const bf16* brow = B + (size_t)(r0 + col) * K + hi * 8;

  v8f acc[4] = {};
  BV16 a_cur, b_cur[4];

  // prologue: k = 0
  a_cur.h[0] = *(const v8bf*)(arow);
  a_cur.h[1] = *(const v8bf*)(arow + 16);
#pragma unroll
  for (int n = 0; n < 4; ++n) {
    const bf16* bp = brow + (size_t)n * 16 * K;
    b_cur[n].h[0] = *(const v8bf*)bp;
    b_cur[n].h[1] = *(const v8bf*)(bp + 16);
  }

  for (int k = 32; k < K; k += 32) {
    // issue next iteration's loads first (independent of current WMMAs)
    BV16 a_nxt, b_nxt[4];
    a_nxt.h[0] = *(const v8bf*)(arow + k);
    a_nxt.h[1] = *(const v8bf*)(arow + k + 16);
#pragma unroll
    for (int n = 0; n < 4; ++n) {
      const bf16* bp = brow + (size_t)n * 16 * K + k;
      b_nxt[n].h[0] = *(const v8bf*)bp;
      b_nxt[n].h[1] = *(const v8bf*)(bp + 16);
    }
#pragma unroll
    for (int n = 0; n < 4; ++n)
      acc[n] = wmma_bf16(a_cur.v, b_cur[n].v, acc[n]);
    a_cur = a_nxt;
#pragma unroll
    for (int n = 0; n < 4; ++n) b_cur[n] = b_nxt[n];
  }
  // epilogue
#pragma unroll
  for (int n = 0; n < 4; ++n)
    acc[n] = wmma_bf16(a_cur.v, b_cur[n].v, acc[n]);

#pragma unroll
  for (int n = 0; n < 4; ++n) {
    const int rc = r0 + n * 16 + col;
    const float bv = bias[rc];
    float* crow = C + (size_t)(t0 + hi * 8) * N + rc;
#pragma unroll
    for (int v = 0; v < 8; ++v) crow[(size_t)v * N] = acc[n][v] + bv;
  }
}

// ---------------------------------------------------------------------------
// Persistent LSTM scan. 128 blocks x 256 threads (8 wave32s).
// Block b owns h-elements j in [16b, 16b+16); its 64 W_hh rows (i/f/g/o x 16)
// stay resident in LDS as bf16 (256 KB of CDNA5's 320 KB WGP LDS).
// Cross-block sync per step uses distributed release/acquire flags (one
// padded flag per block, polled by 128 threads in parallel) instead of a
// serialized central atomic barrier.
// Double-buffer safety: block b enters step t only after all flags >= t,
// which implies every block finished its step t-1 reads of hbuf[(t+1)&1]
// before anyone writes that buffer in step t.
// ---------------------------------------------------------------------------
__global__ __launch_bounds__(256) void lstm_scan_kernel(
    const float* __restrict__ Whh,  // [G4, HID] f32 (converted to bf16 on load)
    const float* __restrict__ xg,   // [TT, G4]  f32 gate pre-activations
    float* __restrict__ hbuf,       // [2, HID]  double-buffered hidden state
    bf16* __restrict__ hsbf,        // [TT, HID] hidden trace (bf16) for out GEMM
    int* __restrict__ flags)        // [SCAN_BLOCKS * FLAG_STRIDE]
{
  __shared__ bf16  shW[4 * 16 * HID];   // 256 KB: rows (g*16+m), contiguous K
  __shared__ bf16  shh[HID];            // 4 KB: h_prev in bf16
  __shared__ float parts[8][4][16];     // per-wave partial sums
  __shared__ float carr[16];            // cell state (this block's slice)

  const int tid  = threadIdx.x;
  const int lane = tid & 31;
  const int wave = tid >> 5;
  const int col  = lane & 15;
  const int hi   = lane >> 4;
  const int j0   = blockIdx.x * 16;

  // Load + convert this block's 64 W_hh rows into LDS (once).
  for (int idx = tid; idx < 4 * 16 * HID; idx += 256) {
    int row = idx / HID;              // g*16 + m
    int k   = idx - row * HID;
    int g   = row >> 4;
    int m   = row & 15;
    shW[idx] = (bf16)Whh[(size_t)(g * HID + j0 + m) * HID + k];
  }
  if (tid < 16) carr[tid] = 0.0f;
  __syncthreads();

  const int kw = wave * 256;            // this wave's K slice (8 chunks of 32)

  for (int t = 0; t < TT; ++t) {
    const float* hprev = hbuf + (t & 1) * HID;
    for (int i = tid; i < HID; i += 256) shh[i] = (bf16)hprev[i];

    // Issue this step's xg gate loads early; latency hides under the WMMAs.
    float xgv[4];
    if (tid < 16) {
#pragma unroll
      for (int g = 0; g < 4; ++g)
        xgv[g] = xg[(size_t)t * G4 + g * HID + j0 + tid];
    }
    __syncthreads();

    v8f acc[4] = {};
    for (int kc = 0; kc < 256; kc += 32) {
      const int k = kw + kc;
      // B: broadcast h chunk into every column (all lanes load same halves;
      // conflict-free LDS broadcast, no EXEC divergence around the WMMA).
      BV16 b;
      b.h[0] = *(const v8bf*)&shh[k + hi * 8];
      b.h[1] = *(const v8bf*)&shh[k + hi * 8 + 16];
#pragma unroll
      for (int g = 0; g < 4; ++g) {
        const bf16* ar = &shW[(size_t)(g * 16 + col) * HID + k + hi * 8];
        BV16 a;
        a.h[0] = *(const v8bf*)ar;
        a.h[1] = *(const v8bf*)(ar + 16);
        acc[g] = wmma_bf16(a.v, b.v, acc[g]);
      }
    }

    // Column 0 of C = A*h: lane 0 carries M=0..7 (v0..7), lane 16 M=8..15.
    if (col == 0) {
      const int mb = hi * 8;
#pragma unroll
      for (int g = 0; g < 4; ++g)
#pragma unroll
        for (int v = 0; v < 8; ++v) parts[wave][g][mb + v] = acc[g][v];
    }
    __syncthreads();

    if (tid < 16) {
      const int m = tid;
      float s[4];
#pragma unroll
      for (int g = 0; g < 4; ++g) {
        float a = xgv[g];
#pragma unroll
        for (int w = 0; w < 8; ++w) a += parts[w][g][m];
        s[g] = a;
      }
      if (t + 1 < TT) {
#pragma unroll
        for (int g = 0; g < 4; ++g)
          __builtin_prefetch(&xg[(size_t)(t + 1) * G4 + g * HID + j0 + m], 0, 3);
      }
      float iv = 1.0f / (1.0f + __expf(-s[0]));
      float fv = 1.0f / (1.0f + __expf(-s[1]));
      float gv = tanhf(s[2]);
      float ov = 1.0f / (1.0f + __expf(-s[3]));
      float c  = fv * carr[m] + iv * gv;
      carr[m]  = c;
      float h  = ov * tanhf(c);
      hbuf[((t + 1) & 1) * HID + j0 + m] = h;
      hsbf[(size_t)t * HID + j0 + m]     = (bf16)h;
    }

    // Publish h_{t+1}: fence block's writes device-wide, release own flag.
    __threadfence();
    __syncthreads();
    if (tid == 0)
      __hip_atomic_store(&flags[blockIdx.x * FLAG_STRIDE], t + 1,
                         __ATOMIC_RELEASE, __HIP_MEMORY_SCOPE_AGENT);
    // Wait (in parallel, one flag per thread) until every block published.
    if (tid < SCAN_BLOCKS) {
      const int* fp = flags + tid * FLAG_STRIDE;
      while (__hip_atomic_load(fp, __ATOMIC_ACQUIRE, __HIP_MEMORY_SCOPE_AGENT) < t + 1)
        __builtin_amdgcn_s_sleep(1);
    }
    __syncthreads();
  }
}

// ---------------------------------------------------------------------------
// Launch
// ---------------------------------------------------------------------------
extern "C" void kernel_launch(void* const* d_in, const int* in_sizes, int n_in,
                              void* d_out, int out_size, void* d_ws, size_t ws_size,
                              hipStream_t stream) {
  (void)in_sizes; (void)n_in; (void)out_size; (void)ws_size;

  const float* X    = (const float*)d_in[0];  // [TT, EMB]
  const float* Wih  = (const float*)d_in[1];  // [G4, EMB]
  const float* Whh  = (const float*)d_in[2];  // [G4, HID]
  const float* bih  = (const float*)d_in[3];  // [G4]
  const float* bhh  = (const float*)d_in[4];  // [G4]
  const float* Wlin = (const float*)d_in[5];  // [OUTD, HID]
  const float* blin = (const float*)d_in[6];  // [OUTD]
  float* out = (float*)d_out;                 // [TT, OUTD]

  char* base = (char*)d_ws;
  size_t off = 0;
  auto take = [&](size_t bytes) -> void* {
    void* p = base + off;
    off += (bytes + 255) & ~(size_t)255;
    return p;
  };
  float* xg     = (float*)take((size_t)TT * G4 * sizeof(float));   // 256 MB
  bf16*  Xbf    = (bf16*) take((size_t)TT * EMB * sizeof(bf16));   // 16 MB
  bf16*  Wihbf  = (bf16*) take((size_t)G4 * EMB * sizeof(bf16));   // 16 MB
  bf16*  Wlinbf = (bf16*) take((size_t)OUTD * HID * sizeof(bf16)); // 4 MB
  bf16*  hsbf   = (bf16*) take((size_t)TT * HID * sizeof(bf16));   // 32 MB
  float* bias   = (float*)take((size_t)G4 * sizeof(float));
  float* hbuf   = (float*)take((size_t)2 * HID * sizeof(float));
  int*   flags  = (int*)  take((size_t)SCAN_BLOCKS * FLAG_STRIDE * sizeof(int));

  // 1) convert + bias + zero state/flags
  prep_kernel<<<2048, 256, 0, stream>>>(X, Wih, bih, bhh, Wlin,
                                        Xbf, Wihbf, Wlinbf, bias, hbuf, flags);
  // 2) x_gates = X @ W_ih^T + (b_ih + b_hh)   [TT, G4]
  gemm_bf16_kernel<<<dim3(TT / 16, G4 / 512), 256, 0, stream>>>(
      Xbf, Wihbf, bias, xg, EMB, G4);
  // 3) sequential LSTM scan (persistent, LDS-resident W_hh, flag handshake)
  lstm_scan_kernel<<<SCAN_BLOCKS, 256, 0, stream>>>(Whh, xg, hbuf, hsbf, flags);
  // 4) out = hs @ W_lin^T + b_lin   [TT, OUTD]
  gemm_bf16_kernel<<<dim3(TT / 16, OUTD / 512), 256, 0, stream>>>(
      hsbf, Wlinbf, blin, out, HID, OUTD);
}
